// CausalSelfAttention_84696755077715
// MI455X (gfx1250) — matrix operand order repaired and checked by reading, not compiled, not run on hardware
//
#include <hip/hip_runtime.h>
#include <hip/hip_bf16.h>

// ---------------------------------------------------------------------------
// Causal self-attention for MI455X (gfx1250), bf16 WMMA pipeline.
//   1) f32->bf16 convert x; transpose-convert Wqkv, Wproj (weights packed [N,K])
//   2) QKV GEMM (WMMA bf16) -> Q(scaled)/K [B,NH,T,HD], V transposed [B,NH,HD,T]
//   3) Flash attention: double-buffered async K/V staging, online softmax
//   4) Projection GEMM (WMMA bf16 -> f32)
// All LDS staging is b128 async-to-LDS; all fragment reads are b128.
// ---------------------------------------------------------------------------

typedef __bf16 bf16_t;
typedef __attribute__((ext_vector_type(16))) __bf16 v16bf;
typedef __attribute__((ext_vector_type(8)))  __bf16 v8bf;
typedef __attribute__((ext_vector_type(8)))  float  v8f;
typedef __attribute__((ext_vector_type(4)))  int    i32x4;

#define B_SZ   2
#define T_SZ   2048
#define C_SZ   768
#define NH_SZ  12
#define HD_SZ  64
#define N3C    (3 * C_SZ)          // 2304
#define MROWS  (B_SZ * T_SZ)       // 4096

// ---------------------------------------------------------------------------
// Async global->LDS copy (CDNA5 GLOBAL_LOAD_ASYNC_TO_LDS_B128, ASYNCcnt)
// ---------------------------------------------------------------------------
#if defined(__has_builtin)
#  if __has_builtin(__builtin_amdgcn_global_load_async_to_lds_b128)
#    define HAVE_ASYNC_LDS 1
#  endif
#endif
#ifndef HAVE_ASYNC_LDS
#  define HAVE_ASYNC_LDS 0
#endif

typedef __attribute__((address_space(1))) i32x4 gas_i32x4;
typedef __attribute__((address_space(3))) i32x4 las_i32x4;

__device__ __forceinline__ void cp16(bf16_t* l, const bf16_t* g) {
#if HAVE_ASYNC_LDS
    // generic->AS pointers via integer round trip (LDS generic low 32 bits = offset)
    __builtin_amdgcn_global_load_async_to_lds_b128(
        (gas_i32x4*)(uintptr_t)(const void*)g,
        (las_i32x4*)(unsigned int)(uintptr_t)(void*)l,
        0, 0);
#else
    *(v8bf*)l = *(const v8bf*)g;
#endif
}

template <int N>
__device__ __forceinline__ void async_wait() {
#if HAVE_ASYNC_LDS
    asm volatile("s_wait_asynccnt %0" :: "i"(N) : "memory");
#endif
}

// ---------------------------------------------------------------------------
// Fragment helpers (contiguous k order k = 16h + e, same for A and B)
// ---------------------------------------------------------------------------
__device__ __forceinline__ v16bf pack16(v8bf lo, v8bf hi) {
    v16bf r;
#pragma unroll
    for (int i = 0; i < 8; ++i) { r[i] = lo[i]; r[i + 8] = hi[i]; }
    return r;
}

// A-matrix 16x32 bf16: lane half h; elems e -> k = h*16 + e (16 contiguous)
__device__ __forceinline__ v16bf ld_frag_a(const bf16_t* rowptr, int h) {
    v8bf lo = *(const v8bf*)(rowptr + h * 16);
    v8bf hi = *(const v8bf*)(rowptr + h * 16 + 8);
    return pack16(lo, hi);
}

// B-matrix 32x16 bf16: lane half h; elems e -> k = h*16 + e (16 contiguous)
__device__ __forceinline__ v16bf ld_frag_b(const bf16_t* p) {
    v8bf lo = *(const v8bf*)(p);
    v8bf hi = *(const v8bf*)(p + 8);
    return pack16(lo, hi);
}

__device__ __forceinline__ v8f wmma_bf16(v16bf a, v16bf b, v8f c) {
    return __builtin_amdgcn_wmma_f32_16x16x32_bf16(
        false, a, false, b, (short)0, c, false, false);
}

// ---------------------------------------------------------------------------
// Conversion kernels
// ---------------------------------------------------------------------------
__global__ void cvt_f32_to_bf16(const float* __restrict__ s,
                                bf16_t* __restrict__ d, int n) {
    int i = blockIdx.x * blockDim.x + threadIdx.x;
    int stride = gridDim.x * blockDim.x;
    for (; i < n; i += stride) d[i] = (bf16_t)s[i];
}

// src [R, N] f32 -> dst [N, R] bf16 (weight packing: B operand becomes [N,K])
__global__ void cvt_transpose_bf16(const float* __restrict__ s,
                                   bf16_t* __restrict__ d, int R, int N) {
    int i = blockIdx.x * blockDim.x + threadIdx.x;
    int stride = gridDim.x * blockDim.x;
    int total = R * N;
    for (; i < total; i += stride) {
        int r = i / N, n = i - r * N;
        d[(size_t)n * R + r] = (bf16_t)s[i];
    }
}

// ---------------------------------------------------------------------------
// GEMM core: C[128x128] tile, BK=32, 8 waves (4m x 2n), wave does 32x64.
// A [M,K] row-major, Bt [N,K] row-major (pre-transposed weights).
// Double-buffered LDS, async staging.
// ---------------------------------------------------------------------------
#define LSTRIDE 40                 // 32 + 8 pad (80B rows, 16B aligned)
#define GBUF    (128 * LSTRIDE)    // elems per buffer

__device__ __forceinline__ void gemm_core(
    const bf16_t* __restrict__ A, const bf16_t* __restrict__ Bt,
    int Kdim, int m0, int n0, bf16_t* ldsA, bf16_t* ldsB, v8f acc[2][4])
{
    const int tid  = threadIdx.x;
    const int lane = tid & 31;
    const int wid  = tid >> 5;
    const int ln   = lane & 15;
    const int lh   = lane >> 4;
    const int wm   = wid & 3;
    const int wn   = wid >> 2;

    // staging addresses: thread copies 16 contiguous bf16 from each operand
    const int trow = tid >> 1;
    const int tcol = (tid & 1) * 16;
    const bf16_t* gA = A + (size_t)(m0 + trow) * Kdim + tcol;
    const bf16_t* gB = Bt + (size_t)(n0 + trow) * Kdim + tcol;
    bf16_t* lA = ldsA + trow * LSTRIDE + tcol;
    bf16_t* lB = ldsB + trow * LSTRIDE + tcol;

    auto issue = [&](int k0, int buf) {
        cp16(lA + buf * GBUF,     gA + k0);
        cp16(lA + buf * GBUF + 8, gA + k0 + 8);
        cp16(lB + buf * GBUF,     gB + k0);
        cp16(lB + buf * GBUF + 8, gB + k0 + 8);
    };

    const int nsteps = Kdim / 32;
    issue(0, 0);
    for (int s = 0; s < nsteps; ++s) {
        const int buf = s & 1;
        __syncthreads();                       // WAR: buf^1 reads finished
        if (s + 1 < nsteps) { issue((s + 1) * 32, buf ^ 1); async_wait<4>(); }
        else                {                               async_wait<0>(); }
        __syncthreads();                       // all threads' cur-tile copies done

        const bf16_t* ca = ldsA + buf * GBUF;
        const bf16_t* cb = ldsB + buf * GBUF;
        v16bf af[2], bfr[4];
#pragma unroll
        for (int tm = 0; tm < 2; ++tm)
            af[tm] = ld_frag_a(&ca[(wm * 32 + tm * 16 + ln) * LSTRIDE], lh);
#pragma unroll
        for (int tn = 0; tn < 4; ++tn)
            bfr[tn] = ld_frag_b(&cb[(wn * 64 + tn * 16 + ln) * LSTRIDE + lh * 16]);
#pragma unroll
        for (int tm = 0; tm < 2; ++tm)
#pragma unroll
            for (int tn = 0; tn < 4; ++tn)
                acc[tm][tn] = wmma_bf16(af[tm], bfr[tn], acc[tm][tn]);
    }
}

// ---------------------------------------------------------------------------
// Kernel 2: QKV GEMM. xb[4096,768] x WqkvT[2304,768] -> Q/K [B,NH,T,HD],
// V transposed [B,NH,HD,T]. Q pre-scaled by 1/sqrt(64).
// ---------------------------------------------------------------------------
__global__ __launch_bounds__(256) void qkv_gemm_kernel(
    const bf16_t* __restrict__ xb, const bf16_t* __restrict__ wt,
    bf16_t* __restrict__ Qo, bf16_t* __restrict__ Ko, bf16_t* __restrict__ Vo)
{
    __shared__ __align__(16) bf16_t ldsA[2 * GBUF];
    __shared__ __align__(16) bf16_t ldsB[2 * GBUF];

    const int m0 = blockIdx.y * 128;
    const int n0 = blockIdx.x * 128;
    const int lane = threadIdx.x & 31;
    const int wid  = threadIdx.x >> 5;
    const int ln = lane & 15, lh = lane >> 4;
    const int wm = wid & 3, wn = wid >> 2;

    v8f acc[2][4];
#pragma unroll
    for (int i = 0; i < 2; ++i)
#pragma unroll
        for (int j = 0; j < 4; ++j) acc[i][j] = (v8f)0.0f;

    gemm_core(xb, wt, C_SZ, m0, n0, ldsA, ldsB, acc);

#pragma unroll
    for (int tm = 0; tm < 2; ++tm)
#pragma unroll
        for (int tn = 0; tn < 4; ++tn)
#pragma unroll
            for (int r = 0; r < 8; ++r) {
                int m  = m0 + wm * 32 + tm * 16 + r + 8 * lh;
                int nc = n0 + wn * 64 + tn * 16 + ln;
                float v = acc[tm][tn][r];
                int which = nc / C_SZ;
                int inner = nc - which * C_SZ;
                int head = inner >> 6;
                int d    = inner & 63;
                int bb = m >> 11;
                int t  = m & (T_SZ - 1);
                if (which == 0) {
                    Qo[(((size_t)bb * NH_SZ + head) * T_SZ + t) * HD_SZ + d] =
                        (bf16_t)(v * 0.125f);
                } else if (which == 1) {
                    Ko[(((size_t)bb * NH_SZ + head) * T_SZ + t) * HD_SZ + d] =
                        (bf16_t)v;
                } else {
                    Vo[(((size_t)bb * NH_SZ + head) * HD_SZ + d) * T_SZ + t] =
                        (bf16_t)v;   // transposed: [b,h,d,t]
                }
            }
}

// ---------------------------------------------------------------------------
// Kernel 3: flash attention. Block = (b,h, 128-query tile); 8 waves x 16 rows.
// K tiles [128key x 64d], V tiles already transposed [64d x 128key]; both
// double-buffered via async-to-LDS. P round-trips through wave-private LDS.
// ---------------------------------------------------------------------------
#define KSTRIDE 72                  // 64 + 8
#define VSTRIDE 136                 // 128 + 8
#define PSTRIDE 136
#define KBUF (128 * KSTRIDE)
#define VBUF (HD_SZ * VSTRIDE)
#define PBUF (16 * PSTRIDE)

__global__ __launch_bounds__(256) void flash_attn_kernel(
    const bf16_t* __restrict__ Qb, const bf16_t* __restrict__ Kb,
    const bf16_t* __restrict__ Vtb, bf16_t* __restrict__ Ob)
{
    extern __shared__ __align__(16) char smem[];
    bf16_t* ldsK  = (bf16_t*)smem;          // 2 buffers
    bf16_t* ldsVt = ldsK + 2 * KBUF;        // 2 buffers
    bf16_t* ldsP  = ldsVt + 2 * VBUF;       // 8 waves x 16 x PSTRIDE

    const int nqb = T_SZ / 128;
    const int bh  = blockIdx.x / nqb;
    const int qb  = blockIdx.x % nqb;
    const int b   = bh / NH_SZ;
    const int h   = bh % NH_SZ;

    const int tid  = threadIdx.x;
    const int lane = tid & 31;
    const int wid  = tid >> 5;
    const int ln   = lane & 15;
    const int lh   = lane >> 4;

    const size_t headoff = ((size_t)(b * NH_SZ + h)) * T_SZ * HD_SZ;
    const bf16_t* Qh  = Qb + headoff;
    const bf16_t* Kh  = Kb + headoff;
    const bf16_t* Vth = Vtb + headoff;      // [d][t] rows of length T

    const int q0 = qb * 128 + wid * 16;

    // Q fragments (registers, straight from global; rows 128B-aligned)
    v16bf aQ[2];
#pragma unroll
    for (int kc = 0; kc < 2; ++kc)
        aQ[kc] = ld_frag_a(Qh + (size_t)(q0 + ln) * HD_SZ + kc * 32, lh);

    // staging addresses
    const int krow = tid >> 1, kcol = (tid & 1) * 32;       // K: [key][d]
    const bf16_t* gK = Kh + (size_t)krow * HD_SZ + kcol;
    bf16_t* lK = ldsK + krow * KSTRIDE + kcol;
    const int vrow = tid >> 2, vcol = (tid & 3) * 32;       // Vt: [d][key]
    const bf16_t* gV = Vth + (size_t)vrow * T_SZ + vcol;
    bf16_t* lV = ldsVt + vrow * VSTRIDE + vcol;

    auto issue = [&](int jb, int buf) {
        const int k0 = jb * 128;
#pragma unroll
        for (int i = 0; i < 4; ++i)
            cp16(lK + buf * KBUF + i * 8, gK + (size_t)k0 * HD_SZ + i * 8);
#pragma unroll
        for (int i = 0; i < 4; ++i)
            cp16(lV + buf * VBUF + i * 8, gV + k0 + i * 8);
    };

    float rm[8], rl[8];
    v8f oacc[4];
#pragma unroll
    for (int r = 0; r < 8; ++r) { rm[r] = -1.0e30f; rl[r] = 0.0f; }
#pragma unroll
    for (int on = 0; on < 4; ++on) oacc[on] = (v8f)0.0f;

    bf16_t* myP = ldsP + wid * PBUF;

    issue(0, 0);
    for (int jb = 0; jb <= qb; ++jb) {
        const int buf = jb & 1;
        const int k0  = jb * 128;
        __syncthreads();                          // WAR on buffer about to fill
        if (jb < qb) { issue(jb + 1, buf ^ 1); async_wait<8>(); }
        else         {                         async_wait<0>(); }
        __syncthreads();                          // cur K/V tiles resident

        const bf16_t* kb = ldsK + buf * KBUF;
        const bf16_t* vb = ldsVt + buf * VBUF;

        // ---- S = Q K^T : 8 n-tiles x 2 k-steps ----
        v8f sacc[8];
#pragma unroll
        for (int tn = 0; tn < 8; ++tn) sacc[tn] = (v8f)0.0f;
#pragma unroll
        for (int tn = 0; tn < 8; ++tn)
#pragma unroll
            for (int kc = 0; kc < 2; ++kc) {
                v16bf bk = ld_frag_b(&kb[(tn * 16 + ln) * KSTRIDE + kc * 32 + lh * 16]);
                sacc[tn] = wmma_bf16(aQ[kc], bk, sacc[tn]);
            }

        // ---- causal mask (diagonal block only) ----
        if (jb == qb) {
#pragma unroll
            for (int tn = 0; tn < 8; ++tn) {
                int kpos = k0 + tn * 16 + ln;
#pragma unroll
                for (int r = 0; r < 8; ++r)
                    if (kpos > q0 + r + 8 * lh) sacc[tn][r] = -1.0e30f;
            }
        }

        // ---- online softmax over rows (r + 8*lh), 16-lane groups ----
        float mnew[8], scl[8];
#pragma unroll
        for (int r = 0; r < 8; ++r) {
            float mx = sacc[0][r];
#pragma unroll
            for (int tn = 1; tn < 8; ++tn) mx = fmaxf(mx, sacc[tn][r]);
            mx = fmaxf(mx, __shfl_xor(mx, 1, 32));
            mx = fmaxf(mx, __shfl_xor(mx, 2, 32));
            mx = fmaxf(mx, __shfl_xor(mx, 4, 32));
            mx = fmaxf(mx, __shfl_xor(mx, 8, 32));
            mnew[r] = fmaxf(rm[r], mx);
            scl[r]  = __expf(rm[r] - mnew[r]);
            rm[r]   = mnew[r];
        }
#pragma unroll
        for (int tn = 0; tn < 8; ++tn)
#pragma unroll
            for (int r = 0; r < 8; ++r)
                sacc[tn][r] = __expf(sacc[tn][r] - mnew[r]);
#pragma unroll
        for (int r = 0; r < 8; ++r) {
            float s = 0.0f;
#pragma unroll
            for (int tn = 0; tn < 8; ++tn) s += sacc[tn][r];
            s += __shfl_xor(s, 1, 32);
            s += __shfl_xor(s, 2, 32);
            s += __shfl_xor(s, 4, 32);
            s += __shfl_xor(s, 8, 32);
            rl[r] = rl[r] * scl[r] + s;
        }
#pragma unroll
        for (int on = 0; on < 4; ++on)
#pragma unroll
            for (int r = 0; r < 8; ++r) oacc[on][r] *= scl[r];

        // ---- P: C-layout -> wave-private LDS (A-layout source) ----
#pragma unroll
        for (int tn = 0; tn < 8; ++tn)
#pragma unroll
            for (int r = 0; r < 8; ++r)
                myP[(r + 8 * lh) * PSTRIDE + tn * 16 + ln] = (bf16_t)sacc[tn][r];
        asm volatile("s_wait_dscnt 0" ::: "memory");      // wave-local LDS RAW

        // ---- O += P V : 4 j-steps x 4 d-tiles ----
#pragma unroll
        for (int js = 0; js < 4; ++js) {
            v16bf aP = ld_frag_a(myP + ln * PSTRIDE + js * 32, lh);
#pragma unroll
            for (int on = 0; on < 4; ++on) {
                v16bf bv = ld_frag_b(&vb[(on * 16 + ln) * VSTRIDE + js * 32 + lh * 16]);
                oacc[on] = wmma_bf16(aP, bv, oacc[on]);
            }
        }
    }

    // ---- epilogue: normalize, write [B,T,C] bf16 ----
    float rli[8];
#pragma unroll
    for (int r = 0; r < 8; ++r) rli[r] = 1.0f / rl[r];
#pragma unroll
    for (int on = 0; on < 4; ++on)
#pragma unroll
        for (int r = 0; r < 8; ++r) {
            int t = q0 + r + 8 * lh;
            int d = on * 16 + ln;
            Ob[((size_t)b * T_SZ + t) * C_SZ + h * HD_SZ + d] =
                (bf16_t)(oacc[on][r] * rli[r]);
        }
}

// ---------------------------------------------------------------------------
// Kernel 4: projection GEMM. attn[4096,768] x WprojT[768,768] -> f32 out
// ---------------------------------------------------------------------------
__global__ __launch_bounds__(256) void proj_gemm_kernel(
    const bf16_t* __restrict__ ab, const bf16_t* __restrict__ wt,
    float* __restrict__ out)
{
    __shared__ __align__(16) bf16_t ldsA[2 * GBUF];
    __shared__ __align__(16) bf16_t ldsB[2 * GBUF];

    const int m0 = blockIdx.y * 128;
    const int n0 = blockIdx.x * 128;
    const int lane = threadIdx.x & 31;
    const int wid  = threadIdx.x >> 5;
    const int ln = lane & 15, lh = lane >> 4;
    const int wm = wid & 3, wn = wid >> 2;

    v8f acc[2][4];
#pragma unroll
    for (int i = 0; i < 2; ++i)
#pragma unroll
        for (int j = 0; j < 4; ++j) acc[i][j] = (v8f)0.0f;

    gemm_core(ab, wt, C_SZ, m0, n0, ldsA, ldsB, acc);

#pragma unroll
    for (int tm = 0; tm < 2; ++tm)
#pragma unroll
        for (int tn = 0; tn < 4; ++tn)
#pragma unroll
            for (int r = 0; r < 8; ++r) {
                int m  = m0 + wm * 32 + tm * 16 + r + 8 * lh;
                int nc = n0 + wn * 64 + tn * 16 + ln;
                out[(size_t)m * C_SZ + nc] = acc[tm][tn][r];
            }
}

// ---------------------------------------------------------------------------
// Host launch
// ---------------------------------------------------------------------------
extern "C" void kernel_launch(void* const* d_in, const int* in_sizes, int n_in,
                              void* d_out, int out_size, void* d_ws, size_t ws_size,
                              hipStream_t stream) {
    const float* x     = (const float*)d_in[0];
    // d_in[1] = causal mask (recomputed analytically in-kernel; unused)
    const float* Wqkv  = (const float*)d_in[2];
    const float* Wproj = (const float*)d_in[3];
    float* out = (float*)d_out;

    const int xN     = B_SZ * T_SZ * C_SZ;            // 3,145,728
    const int wqkvN  = C_SZ * N3C;                    // 1,769,472
    const int wprojN = C_SZ * C_SZ;                   //   589,824
    const int headN  = B_SZ * NH_SZ * T_SZ * HD_SZ;   // 3,145,728

    char* w = (char*)d_ws;
    auto take = [&](size_t elems) {
        bf16_t* p = (bf16_t*)w;
        w += (elems * sizeof(bf16_t) + 255) & ~(size_t)255;
        return p;
    };
    bf16_t* xb      = take(xN);
    bf16_t* wqkvT   = take(wqkvN);   // [2304, 768]
    bf16_t* wprojT  = take(wprojN);  // [768, 768]
    bf16_t* Qb      = take(headN);   // [b,h,t,d]
    bf16_t* Kb      = take(headN);   // [b,h,t,d]
    bf16_t* Vtb     = take(headN);   // [b,h,d,t]
    bf16_t* attnb   = take(xN);      // [b,t,c]

    cvt_f32_to_bf16  <<<2048, 256, 0, stream>>>(x, xb, xN);
    cvt_transpose_bf16<<<2048, 256, 0, stream>>>(Wqkv,  wqkvT,  C_SZ, N3C);
    cvt_transpose_bf16<<<1024, 256, 0, stream>>>(Wproj, wprojT, C_SZ, C_SZ);

    qkv_gemm_kernel<<<dim3(N3C / 128, MROWS / 128), 256, 0, stream>>>(
        xb, wqkvT, Qb, Kb, Vtb);

    const size_t flash_lds =
        (2 * KBUF + 2 * VBUF + 8 * PBUF) * sizeof(bf16_t);   // 106,496 B
    flash_attn_kernel<<<B_SZ * NH_SZ * (T_SZ / 128), 256, flash_lds, stream>>>(
        Qb, Kb, Vtb, attnb);

    proj_gemm_kernel<<<dim3(C_SZ / 128, MROWS / 128), 256, 0, stream>>>(
        attnb, wprojT, out);
}
